// FragGNNSmall_4432406249779
// MI455X (gfx1250) — compile-verified, matrix operand order checked by reading, not run on hardware
//
#include <hip/hip_runtime.h>
#include <hip/hip_bf16.h>

#define HD 128

typedef __attribute__((ext_vector_type(16))) __bf16 v16bf;
typedef __attribute__((ext_vector_type(8)))  float  v8f;

// ---------------------------------------------------------------------------
// Fused WMMA GEMM:  C = act( A[gather] @ W + bias ), optional scatter-add.
//   A: M x KT f32, W: KT x 128 f32 (LDS-resident, bf16, pre-swizzled into
//   WMMA fragment order), C: M x 128 (or scatter target via sidx).
// MODE: 0 = store, 1 = relu+store, 2 = relu + atomic scatter-add (seg_sum).
// KT is compile-time (16/32/128) -> fully unrolled, branchless K-loop.
// Block = 256 threads = 8 waves; wave w owns rows [base+16w, base+16w+16).
// Rows past M are clamped to row 0 on the load side; stores take a
// wave-uniform full-tile fast path (unconditional, immediate offsets) and
// only the final partial tile uses per-row guards.
// ---------------------------------------------------------------------------
template<int MODE, bool GATHER, int KT>
__global__ __launch_bounds__(256)
void wmma_gemm128(const float* __restrict__ A, const int* __restrict__ gidx,
                  const float* __restrict__ W, const float* __restrict__ bias,
                  float* __restrict__ C, const int* __restrict__ sidx,
                  int M)
{
    constexpr int KPAD   = (KT + 31) & ~31;
    constexpr int NCHUNK = KPAD / 32;

    // swizzled weights: element ((c*8 + t)*32 + L)*16 + e  holds
    // W[k][n] with k = c*32 + (e>>3)*16 + (L>>4)*8 + (e&7), n = t*16 + (L&15)
    __shared__ __attribute__((aligned(32))) __bf16 sW[NCHUNK * 4096];

    const int tid      = threadIdx.x;
    const int wave     = tid >> 5;
    const int lane     = tid & 31;
    const int laneHalf = lane >> 4;
    const int lrow     = lane & 15;

    if (KT < KPAD) {                        // compile-time; K=16 only
        for (int d = tid; d < NCHUNK * 4096; d += 256) sW[d] = (__bf16)0.0f;
        __syncthreads();
    }
    // source-order fill: coalesced b32 reads of W, scattered b16 LDS stores
    for (int i = tid; i < KT * HD; i += 256) {
        int k = i >> 7, n = i & 127;
        int kk = k & 31, c = k >> 5;
        int e  = ((kk >> 4) << 3) | (kk & 7);
        int L  = (((kk >> 3) & 1) << 4) | (n & 15);
        int t  = n >> 4;
        sW[(((((c << 3) + t) << 5) + L) << 4) + e] = (__bf16)W[i];
    }
    __syncthreads();

    // per-lane bias values (N = t*16 + lrow)
    float bv[8];
#pragma unroll
    for (int t = 0; t < 8; ++t) bv[t] = bias[t * 16 + lrow];

    const int mbase = (blockIdx.x * 8 + wave) * 16;
    const int m0    = mbase + lrow;
    const int mi    = m0 < M ? m0 : 0;             // clamp: loads always valid
    const long row  = GATHER ? (long)gidx[mi] : (long)mi;
    const float* __restrict__ Arow = A + row * (long)KT;

    v8f acc[8];
#pragma unroll
    for (int t = 0; t < 8; ++t) acc[t] = v8f{};

#pragma unroll
    for (int c = 0; c < NCHUNK; ++c) {
        const int kc = c * 32;
        // ---- A fragment: unconditional 16B global loads, cvt to bf16 ----
        v16bf af;
#pragma unroll
        for (int g = 0; g < 2; ++g) {
            const int kb = kc + (g << 4);
            if (kb < KT) {
                float4 p0 = *(const float4*)(Arow + kb + (laneHalf << 3));
                float4 p1 = *(const float4*)(Arow + kb + (laneHalf << 3) + 4);
                af[g * 8 + 0] = (__bf16)p0.x; af[g * 8 + 1] = (__bf16)p0.y;
                af[g * 8 + 2] = (__bf16)p0.z; af[g * 8 + 3] = (__bf16)p0.w;
                af[g * 8 + 4] = (__bf16)p1.x; af[g * 8 + 5] = (__bf16)p1.y;
                af[g * 8 + 6] = (__bf16)p1.z; af[g * 8 + 7] = (__bf16)p1.w;
            } else {                                // compile-time (K=16 tail)
                af[g * 8 + 0] = (__bf16)0.0f; af[g * 8 + 1] = (__bf16)0.0f;
                af[g * 8 + 2] = (__bf16)0.0f; af[g * 8 + 3] = (__bf16)0.0f;
                af[g * 8 + 4] = (__bf16)0.0f; af[g * 8 + 5] = (__bf16)0.0f;
                af[g * 8 + 6] = (__bf16)0.0f; af[g * 8 + 7] = (__bf16)0.0f;
            }
        }
        // ---- 8 column tiles: contiguous 32B fragment loads + WMMA ----
#pragma unroll
        for (int t = 0; t < 8; ++t) {
            v16bf bf = *(const v16bf*)&sW[((((c << 3) + t) << 5) + lane) << 4];
            acc[t] = __builtin_amdgcn_wmma_f32_16x16x32_bf16(
                false, af, false, bf, (short)0, acc[t], false, false);
        }
    }

    // ---- epilogue: C/D layout VGPR r -> (M = r + 8*half, N = lane%16) ----
    const int mtop = mbase + laneHalf * 8;          // first of this half's rows
    if (mbase + 16 <= M) {                          // wave-uniform fast path
        if (MODE == 2) {
            long cr[8];
#pragma unroll
            for (int r = 0; r < 8; ++r) cr[r] = (long)sidx[mtop + r];
#pragma unroll
            for (int t = 0; t < 8; ++t) {
                const int n = t * 16 + lrow;
#pragma unroll
                for (int r = 0; r < 8; ++r)
                    atomicAdd(&C[cr[r] * HD + n], fmaxf(acc[t][r] + bv[t], 0.0f));
            }
        } else {
            float* __restrict__ Cb = C + (long)mtop * HD + lrow;
#pragma unroll
            for (int t = 0; t < 8; ++t) {
#pragma unroll
                for (int r = 0; r < 8; ++r) {
                    float v = acc[t][r] + bv[t];
                    if (MODE == 1) v = fmaxf(v, 0.0f);
                    Cb[r * HD + t * 16] = v;        // immediate offsets
                }
            }
        }
    } else {                                        // partial (last) tile
#pragma unroll
        for (int t = 0; t < 8; ++t) {
            const int n = t * 16 + lrow;
#pragma unroll
            for (int r = 0; r < 8; ++r) {
                int m = mtop + r;
                if (m < M) {
                    float v = acc[t][r] + bv[t];
                    if (MODE >= 1) v = fmaxf(v, 0.0f);
                    if (MODE == 2) atomicAdd(&C[(long)sidx[m] * HD + n], v);
                    else           C[(long)m * HD + n] = v;
                }
            }
        }
    }
}

// ---------------------------------------------------------------------------
// Fused bond-encoder + GINE message + scatter-add:
//   agg[dst[e]] += relu( h[src[e]] + edge_attr[e] @ bondW + bondb )
// bondW column + bias are loop-invariant per thread -> registers, no LDS.
// ---------------------------------------------------------------------------
__global__ __launch_bounds__(256)
void edge_msg_kernel(const float* __restrict__ edge_attr,
                     const int* __restrict__ src, const int* __restrict__ dst,
                     const float* __restrict__ bondW, const float* __restrict__ bondb,
                     const float* __restrict__ h, float* __restrict__ agg, int E)
{
    const int tid = threadIdx.x;
    const int col = tid & 127;
    const int sub = tid >> 7;            // 0..1
    float w[8];
#pragma unroll
    for (int c = 0; c < 8; ++c) w[c] = bondW[c * HD + col];
    const float bb = bondb[col];

    const int ebase = blockIdx.x * 16;
#pragma unroll
    for (int j = 0; j < 16; j += 2) {
        int e = ebase + j + sub;
        if (e < E) {
            float4 ea0 = *(const float4*)(edge_attr + (long)e * 8);
            float4 ea1 = *(const float4*)(edge_attr + (long)e * 8 + 4);
            float ev = bb + ea0.x * w[0] + ea0.y * w[1] + ea0.z * w[2] + ea0.w * w[3]
                          + ea1.x * w[4] + ea1.y * w[5] + ea1.z * w[6] + ea1.w * w[7];
            float m = h[(long)src[e] * HD + col] + ev;
            m = m > 0.0f ? m : 0.0f;
            atomicAdd(&agg[(long)dst[e] * HD + col], m);
        }
    }
}

// z = (1 + eps) * h + agg   (in place into agg), float4
__global__ __launch_bounds__(256)
void z_kernel(float* __restrict__ agg, const float* __restrict__ h,
              const float* __restrict__ eps_ptr, long n)
{
    long t4 = ((long)blockIdx.x * 256 + threadIdx.x) * 4;
    if (t4 >= n) return;
    float s = 1.0f + eps_ptr[0];
    float4 hv = *(const float4*)(h + t4);
    float4 av = *(float4*)(agg + t4);
    av.x = s * hv.x + av.x; av.y = s * hv.y + av.y;
    av.z = s * hv.z + av.z; av.w = s * hv.w + av.w;
    *(float4*)(agg + t4) = av;
}

// per-column sum / sumsq partials -> stats[0..127]=sum, stats[128..255]=sumsq
// thread owns 4 columns; float4 loads; 8-row-group LDS tree reduction.
__global__ __launch_bounds__(256)
void colstats_kernel(const float* __restrict__ X, float* __restrict__ stats, int M)
{
    __shared__ float r0[8][HD];
    __shared__ float r1[8][HD];
    const int tid = threadIdx.x;
    const int c4  = (tid & 31) * 4;
    const int rg  = tid >> 5;                        // 0..7
    float s0 = 0, s1 = 0, s2 = 0, s3 = 0;
    float q0 = 0, q1 = 0, q2 = 0, q3 = 0;
    for (int r = blockIdx.x * 8 + rg; r < M; r += gridDim.x * 8) {
        float4 v = *(const float4*)(X + (long)r * HD + c4);
        s0 += v.x; s1 += v.y; s2 += v.z; s3 += v.w;
        q0 += v.x * v.x; q1 += v.y * v.y; q2 += v.z * v.z; q3 += v.w * v.w;
    }
    r0[rg][c4 + 0] = s0; r0[rg][c4 + 1] = s1; r0[rg][c4 + 2] = s2; r0[rg][c4 + 3] = s3;
    r1[rg][c4 + 0] = q0; r1[rg][c4 + 1] = q1; r1[rg][c4 + 2] = q2; r1[rg][c4 + 3] = q3;
    __syncthreads();
    if (tid < HD) {
        float a = 0, b = 0;
#pragma unroll
        for (int g = 0; g < 8; ++g) { a += r0[g][tid]; b += r1[g][tid]; }
        atomicAdd(&stats[tid],      a);
        atomicAdd(&stats[HD + tid], b);
    }
}

// y = g*(x-m)*rsqrt(v+eps)+b, optional relu; float4 over rows and params
__global__ __launch_bounds__(256)
void bn_relu_kernel(float* __restrict__ X, const float* __restrict__ stats,
                    const float* __restrict__ g, const float* __restrict__ b,
                    int M, int relu)
{
    long t4 = ((long)blockIdx.x * 256 + threadIdx.x) * 4;
    if (t4 >= (long)M * HD) return;
    const int col = (int)(t4 & 127);
    const float invM = 1.0f / (float)M;
    float4 sm = *(const float4*)(stats + col);
    float4 sq = *(const float4*)(stats + HD + col);
    float4 gv = *(const float4*)(g + col);
    float4 bvv= *(const float4*)(b + col);
    float4 xv = *(float4*)(X + t4);
    float m, va;
    m = sm.x * invM; va = sq.x * invM - m * m;
    xv.x = gv.x * (xv.x - m) * rsqrtf(va + 1e-5f) + bvv.x;
    m = sm.y * invM; va = sq.y * invM - m * m;
    xv.y = gv.y * (xv.y - m) * rsqrtf(va + 1e-5f) + bvv.y;
    m = sm.z * invM; va = sq.z * invM - m * m;
    xv.z = gv.z * (xv.z - m) * rsqrtf(va + 1e-5f) + bvv.z;
    m = sm.w * invM; va = sq.w * invM - m * m;
    xv.w = gv.w * (xv.w - m) * rsqrtf(va + 1e-5f) + bvv.w;
    if (relu) {
        xv.x = fmaxf(xv.x, 0.0f); xv.y = fmaxf(xv.y, 0.0f);
        xv.z = fmaxf(xv.z, 0.0f); xv.w = fmaxf(xv.w, 0.0f);
    }
    *(float4*)(X + t4) = xv;
}

// float4 zero with scalar tail
__global__ __launch_bounds__(256)
void zero_kernel(float* __restrict__ p, long n)
{
    long t4 = ((long)blockIdx.x * 256 + threadIdx.x) * 4;
    if (t4 + 4 <= n) {
        *(float4*)(p + t4) = float4{0.f, 0.f, 0.f, 0.f};
    } else {
        for (long i = t4; i < n; ++i) p[i] = 0.0f;
    }
}

__global__ __launch_bounds__(256)
void count_kernel(const int* __restrict__ idx, float* __restrict__ cnt, int M)
{
    int t = blockIdx.x * 256 + threadIdx.x;
    if (t < M) atomicAdd(&cnt[idx[t]], 1.0f);
}

// dst[i,c] += sum[i,c] / max(cnt[i],1) ; float4 (4 cols share one row)
__global__ __launch_bounds__(256)
void add_mean_kernel(float* __restrict__ dst, const float* __restrict__ sum,
                     const float* __restrict__ cnt, int M)
{
    long t4 = ((long)blockIdx.x * 256 + threadIdx.x) * 4;
    if (t4 >= (long)M * HD) return;
    float inv = 1.0f / fmaxf(cnt[t4 >> 7], 1.0f);
    float4 sv = *(const float4*)(sum + t4);
    float4 dv = *(float4*)(dst + t4);
    dv.x += sv.x * inv; dv.y += sv.y * inv;
    dv.z += sv.z * inv; dv.w += sv.w * inv;
    *(float4*)(dst + t4) = dv;
}

// batch pooling scatter: sum[idx[row]] += X[row], count per row once
__global__ __launch_bounds__(256)
void pool_scatter_kernel(const float* __restrict__ X, const int* __restrict__ idx,
                         float* __restrict__ sum, float* __restrict__ cnt, int M)
{
    long t = (long)blockIdx.x * 256 + threadIdx.x;
    if (t >= (long)M * HD) return;
    int row = (int)(t >> 7), col = (int)(t & 127);
    int b   = idx[row];
    atomicAdd(&sum[(long)b * HD + col], X[t]);
    if (col == 0) atomicAdd(&cnt[b], 1.0f);
}

__global__ __launch_bounds__(256)
void mean_div_kernel(const float* __restrict__ sum, const float* __restrict__ cnt,
                     float* __restrict__ out, int B)
{
    long t4 = ((long)blockIdx.x * 256 + threadIdx.x) * 4;
    if (t4 >= (long)B * HD) return;
    float inv = 1.0f / fmaxf(cnt[t4 >> 7], 1.0f);
    float4 sv = *(const float4*)(sum + t4);
    *(float4*)(out + t4) = float4{sv.x * inv, sv.y * inv, sv.z * inv, sv.w * inv};
}

__global__ __launch_bounds__(256)
void combine_relu_kernel(const float* __restrict__ a, const float* __restrict__ b,
                         float* __restrict__ c, long n)
{
    long t4 = ((long)blockIdx.x * 256 + threadIdx.x) * 4;
    if (t4 >= n) return;
    float4 av = *(const float4*)(a + t4);
    float4 bv = *(const float4*)(b + t4);
    *(float4*)(c + t4) = float4{fmaxf(av.x + bv.x, 0.f), fmaxf(av.y + bv.y, 0.f),
                                fmaxf(av.z + bv.z, 0.f), fmaxf(av.w + bv.w, 0.f)};
}

// out[b] = X[b,:] . W[:,0] + bias  (Co == 1, no relu); float4 dot
__global__ __launch_bounds__(256)
void final_out_kernel(const float* __restrict__ X, const float* __restrict__ W,
                      const float* __restrict__ b, float* __restrict__ out, int B)
{
    int t = blockIdx.x * 256 + threadIdx.x;
    if (t >= B) return;
    float acc = b[0];
#pragma unroll 8
    for (int c = 0; c < HD; c += 4) {
        float4 xv = *(const float4*)(X + (long)t * HD + c);
        float4 wv = *(const float4*)(W + c);
        acc += xv.x * wv.x + xv.y * wv.y + xv.z * wv.z + xv.w * wv.w;
    }
    out[t] = acc;
}

// ---------------------------------------------------------------------------
static inline int cdiv(long a, long b) { return (int)((a + b - 1) / b); }

static inline void zero(float* p, long n, hipStream_t s)
{ zero_kernel<<<cdiv(n, 1024), 256, 0, s>>>(p, n); }

static inline void gemm(const float* A, const int* gidx, const float* W,
                        const float* bias, float* C, const int* sidx,
                        int M, int K, int mode, hipStream_t s)
{
    const int g = cdiv(M, 128);
    if (K == 16)
        wmma_gemm128<0, false, 16 ><<<g, 256, 0, s>>>(A, gidx, W, bias, C, sidx, M);
    else if (K == 32)
        wmma_gemm128<0, false, 32 ><<<g, 256, 0, s>>>(A, gidx, W, bias, C, sidx, M);
    else if (mode == 0)
        wmma_gemm128<0, false, 128><<<g, 256, 0, s>>>(A, gidx, W, bias, C, sidx, M);
    else if (mode == 1)
        wmma_gemm128<1, false, 128><<<g, 256, 0, s>>>(A, gidx, W, bias, C, sidx, M);
    else
        wmma_gemm128<2, true,  128><<<g, 256, 0, s>>>(A, gidx, W, bias, C, sidx, M);
}

static inline void bnorm(float* X, float* stats, const float* g, const float* b,
                         int M, int relu, hipStream_t s)
{
    zero(stats, 256, s);
    colstats_kernel<<<512, 256, 0, s>>>(X, stats, M);
    bn_relu_kernel<<<cdiv((long)M * HD, 1024), 256, 0, s>>>(X, stats, g, b, M, relu);
}

extern "C" void kernel_launch(void* const* d_in, const int* in_sizes, int n_in,
                              void* d_out, int out_size, void* d_ws, size_t ws_size,
                              hipStream_t stream)
{
    (void)n_in; (void)ws_size;

    const float* x        = (const float*)d_in[0];
    const float* edge_attr= (const float*)d_in[1];
    const float* frag_attr= (const float*)d_in[2];
    const float* aeW  = (const float*)d_in[3];
    const float* aeb  = (const float*)d_in[4];
    const float* feW  = (const float*)d_in[5];
    const float* feb  = (const float*)d_in[6];
    const float* bondW= (const float*)d_in[7];
    const float* bondb= (const float*)d_in[8];
    const float* eps  = (const float*)d_in[9];
    const float* W1   = (const float*)d_in[10];
    const float* b1   = (const float*)d_in[11];
    const float* bng  = (const float*)d_in[12];
    const float* bnb  = (const float*)d_in[13];
    const float* W2   = (const float*)d_in[14];
    const float* b2   = (const float*)d_in[15];
    const float* abng = (const float*)d_in[16];
    const float* abnb = (const float*)d_in[17];
    const float* fbng = (const float*)d_in[18];
    const float* fbnb = (const float*)d_in[19];
    const float* a2fW = (const float*)d_in[20];
    const float* a2fb = (const float*)d_in[21];
    const float* f2aW = (const float*)d_in[22];
    const float* f2ab = (const float*)d_in[23];
    const float* aoW1 = (const float*)d_in[24];
    const float* aob1 = (const float*)d_in[25];
    const float* aoW2 = (const float*)d_in[26];
    const float* aob2 = (const float*)d_in[27];
    const float* foW1 = (const float*)d_in[28];
    const float* fob1 = (const float*)d_in[29];
    const float* foW2 = (const float*)d_in[30];
    const float* fob2 = (const float*)d_in[31];
    const float* oW1  = (const float*)d_in[32];
    const float* ob1  = (const float*)d_in[33];
    const float* oW2  = (const float*)d_in[34];
    const float* ob2  = (const float*)d_in[35];
    const int* eidx   = (const int*)d_in[36];
    const int* frow   = (const int*)d_in[37];
    const int* fcol   = (const int*)d_in[38];
    const int* batch  = (const int*)d_in[39];
    const int* fbatch = (const int*)d_in[40];

    const int Cin = 16, Ce = 8, Cs = 32;
    const int N  = in_sizes[0] / Cin;
    const int E  = in_sizes[1] / Ce;
    const int Fn = in_sizes[2] / Cs;
    const int L  = in_sizes[9];
    const int Fe = in_sizes[37];
    const int B  = out_size;          // Co == 1 -> out is (B,1)

    const int* src = eidx;
    const int* dst = eidx + E;

    // ---- workspace layout (f32) ----
    float* ws    = (float*)d_ws;
    float* h     = ws;                       // N*128
    float* hf    = h    + (long)N  * HD;     // Fn*128
    float* agg   = hf   + (long)Fn * HD;     // N*128 (agg -> z -> nodesum)
    float* t1    = agg  + (long)N  * HD;     // N*128
    float* fsum  = t1   + (long)N  * HD;     // Fn*128
    float* fcnt  = fsum + (long)Fn * HD;     // Fn    (contiguous with fsum)
    float* ncnt  = fcnt + (long)Fn;          // N
    float* stats = ncnt + (long)N;           // 256
    float* gsum  = stats + 256;              // B*128
    float* gfsum = gsum + (long)B * HD;      // B*128
    float* gcnt  = gfsum + (long)B * HD;     // B   (gsum..gfcnt contiguous)
    float* gfcnt = gcnt + (long)B;           // B
    float* gbuf  = gfcnt + (long)B;          // B*128
    float* gfbuf = gbuf + (long)B * HD;      // B*128
    float* t2    = gfbuf + (long)B * HD;     // B*128
    float* t3    = t2   + (long)B * HD;      // B*128
    float* t3f   = t3   + (long)B * HD;      // B*128

    // ---- encoders ----
    gemm(x,         nullptr, aeW, aeb, h,  nullptr, N,  Cin, 0, stream);
    gemm(frag_attr, nullptr, feW, feb, hf, nullptr, Fn, Cs,  0, stream);

    // ---- message-passing layers ----
    for (int l = 0; l < L; ++l) {
        const long WO = (long)l * HD * HD;
        const long BO = (long)l * HD;

        // GINEConv aggregation (fused bond encoder + relu + scatter-add)
        zero(agg, (long)N * HD, stream);
        edge_msg_kernel<<<cdiv(E, 16), 256, 0, stream>>>(
            edge_attr, src, dst, bondW + (long)l * Ce * HD, bondb + BO, h, agg, E);

        // z = (1+eps)h + agg ; z @ W1 -> BN -> relu ; @ W2 -> BN -> relu
        z_kernel<<<cdiv((long)N * HD, 1024), 256, 0, stream>>>(agg, h, eps + l, (long)N * HD);
        gemm(agg, nullptr, W1 + WO, b1 + BO, t1, nullptr, N, HD, 0, stream);
        bnorm(t1, stats, bng + BO, bnb + BO, N, 1, stream);
        gemm(t1, nullptr, W2 + WO, b2 + BO, h, nullptr, N, HD, 0, stream);
        bnorm(h, stats, abng + BO, abnb + BO, N, 1, stream);

        // atom -> frag inter-message (gather + GEMM + relu + scatter-mean)
        zero(fsum, (long)Fn * HD + Fn, stream);                  // fsum & fcnt
        count_kernel<<<cdiv(Fe, 256), 256, 0, stream>>>(fcol, fcnt, Fe);
        gemm(h, frow, a2fW + WO, a2fb + BO, fsum, fcol, Fe, HD, 2, stream);
        add_mean_kernel<<<cdiv((long)Fn * HD, 1024), 256, 0, stream>>>(hf, fsum, fcnt, Fn);
        bnorm(hf, stats, fbng + BO, fbnb + BO, Fn, 1, stream);

        // frag -> atom inter-message
        zero(agg, (long)N * HD, stream);
        zero(ncnt, N, stream);
        count_kernel<<<cdiv(Fe, 256), 256, 0, stream>>>(frow, ncnt, Fe);
        gemm(hf, fcol, f2aW + WO, f2ab + BO, agg, frow, Fe, HD, 2, stream);
        add_mean_kernel<<<cdiv((long)N * HD, 1024), 256, 0, stream>>>(h, agg, ncnt, N);
    }

    // ---- readout ----
    zero(gsum, 2 * (long)B * HD + 2 * (long)B, stream);          // gsum,gfsum,gcnt,gfcnt
    pool_scatter_kernel<<<cdiv((long)N * HD, 256), 256, 0, stream>>>(h, batch, gsum, gcnt, N);
    pool_scatter_kernel<<<cdiv((long)Fn * HD, 256), 256, 0, stream>>>(hf, fbatch, gfsum, gfcnt, Fn);
    mean_div_kernel<<<cdiv((long)B * HD, 1024), 256, 0, stream>>>(gsum, gcnt, gbuf, B);
    mean_div_kernel<<<cdiv((long)B * HD, 1024), 256, 0, stream>>>(gfsum, gfcnt, gfbuf, B);

    gemm(gbuf, nullptr, aoW1, aob1, t2, nullptr, B, HD, 1, stream);
    gemm(t2,   nullptr, aoW2, aob2, t3, nullptr, B, HD, 1, stream);
    gemm(gfbuf,nullptr, foW1, fob1, t2, nullptr, B, HD, 1, stream);
    gemm(t2,   nullptr, foW2, fob2, t3f,nullptr, B, HD, 1, stream);

    combine_relu_kernel<<<cdiv((long)B * HD, 1024), 256, 0, stream>>>(t3, t3f, gbuf, (long)B * HD);
    gemm(gbuf, nullptr, oW1, ob1, t2, nullptr, B, HD, 1, stream);
    final_out_kernel<<<cdiv(B, 256), 256, 0, stream>>>(t2, oW2, ob2, (float*)d_out, B);
}
